// GAT_67018669687297
// MI455X (gfx1250) — matrix operand order repaired and checked by reading, not compiled, not run on hardware
//
#include <hip/hip_runtime.h>
#include <hip/hip_bf16.h>
#include <math.h>

typedef __attribute__((ext_vector_type(16))) _Float16 v16h;
typedef __attribute__((ext_vector_type(8)))  _Float16 v8h;
typedef __attribute__((ext_vector_type(8)))  float    v8f;

// ---------------------------------------------------------------------------
// order-preserving float <-> uint key (for atomicMax-based segment_max)
// ---------------------------------------------------------------------------
__device__ __forceinline__ unsigned fkey(float f) {
  unsigned b = __float_as_uint(f);
  return (b & 0x80000000u) ? ~b : (b | 0x80000000u);
}
__device__ __forceinline__ float unfkey(unsigned k) {
  unsigned b = (k & 0x80000000u) ? (k ^ 0x80000000u) : ~k;
  return __uint_as_float(b);
}
#define NEG_INF_KEY 0x007FFFFFu  // fkey(-inf)

// ---------------------------------------------------------------------------
// small utility kernels
// ---------------------------------------------------------------------------
__global__ void k_cvt_f16(const float* __restrict__ s, _Float16* __restrict__ d, long n) {
  long i = (long)blockIdx.x * blockDim.x + threadIdx.x;
  if (i < n) d[i] = (_Float16)s[i];
}

__global__ void k_fill_u32(unsigned* __restrict__ p, unsigned v, int n) {
  int i = blockIdx.x * blockDim.x + threadIdx.x;
  if (i < n) p[i] = v;
}

__global__ void k_fill_f32(float* __restrict__ p, float v, long n) {
  long i = (long)blockIdx.x * blockDim.x + threadIdx.x;
  if (i < n) p[i] = v;
}

template <int F>  // F = power of two (64)
__global__ void k_bias_rows(float* __restrict__ out, const float* __restrict__ b, int n) {
  long i = (long)blockIdx.x * blockDim.x + threadIdx.x;
  if (i < (long)n * F) out[i] = b[(int)i & (F - 1)];
}

// Repack W [K, Ncols] f32 -> WMMA B fragments, f16.
// Layout: [ntile][kblock][lane 0..31][16 halfs], where lane<16 covers column
// nt*16+lane, K = kb*32 + 0..7,16..23 ; lane>=16 covers column nt*16+lane-16,
// K = kb*32 + 8..15,24..31 (matches A fragment order).
__global__ void k_prepack_w(const float* __restrict__ W, _Float16* __restrict__ Wp,
                            int K, int Ncols) {
  int idx = blockIdx.x * blockDim.x + threadIdx.x;
  if (idx >= K * Ncols) return;
  int i = idx & 15;
  int L = (idx >> 4) & 31;
  int g = idx >> 9;            // ntile*kblocks + kb
  int kblocks = K >> 5;
  int kb = g % kblocks;
  int nt = g / kblocks;
  int n = nt * 16 + (L & 15);
  int k = kb * 32 + ((L >> 4) << 3) + (i & 7) + ((i >> 3) << 4);
  Wp[idx] = (_Float16)W[k * Ncols + n];
}

// ---------------------------------------------------------------------------
// WMMA GEMM: C[M, NT*16] = A[M, KB*32] * Bprepacked.  One wave per 16-row
// M-tile; A fragments live in registers across all N-tiles.
// A fragment per ISA: lane<16 row M=lane holds K[0..8)+K[16..24);
// lane>=16 holds K[8..16)+K[24..32) of each 32-wide K-block.
// C: vgpr r -> row = 16*tile + r + (lane>=16 ? 8:0), col = lane&15.
// ---------------------------------------------------------------------------
template <int KB, int NT>
__global__ void k_gemm_wmma(const _Float16* __restrict__ A,
                            const _Float16* __restrict__ Bp,
                            float* __restrict__ C, int M) {
  const int K = KB * 32, Ncols = NT * 16;
  int wave = (int)((blockIdx.x * (long)blockDim.x + threadIdx.x) >> 5);
  int lane = threadIdx.x & 31;
  int mtiles = (M + 15) >> 4;
  if (wave >= mtiles) return;

  int row = wave * 16 + (lane & 15);
  if (row >= M) row = M - 1;               // clamp (tail tiles)
  int koff = (lane >> 4) << 3;             // 0 or 8 halfs

  v16h a[KB];
#pragma unroll
  for (int kb = 0; kb < KB; ++kb) {
    const _Float16* p = A + (size_t)row * K + kb * 32 + koff;
    union { v16h v; v8h h[2]; } u;
    u.h[0] = *(const v8h*)(p);
    u.h[1] = *(const v8h*)(p + 16);
    a[kb] = u.v;
  }

  int mstore = wave * 16 + ((lane >> 4) << 3);
  int ncol = lane & 15;
  bool full = (wave * 16 + 16) <= M;       // whole tile in range (common case)
#pragma unroll
  for (int nt = 0; nt < NT; ++nt) {
    v8f acc = {};
#pragma unroll
    for (int kb = 0; kb < KB; ++kb) {
      v16h b = *(const v16h*)(Bp + (((size_t)nt * KB + kb) * 32 + lane) * 16);
      acc = __builtin_amdgcn_wmma_f32_16x16x32_f16(
          false, a[kb], false, b, (short)0, acc, false, false);
    }
    float* o = C + (size_t)mstore * Ncols + nt * 16 + ncol;
    if (full) {
#pragma unroll
      for (int r = 0; r < 8; ++r) o[(size_t)r * Ncols] = acc[r];
    } else {
#pragma unroll
      for (int r = 0; r < 8; ++r)
        if (mstore + r < M) o[(size_t)r * Ncols] = acc[r];
    }
  }
}

// ---------------------------------------------------------------------------
// node-level attention terms: als[i,h] = sum_c h[i,h,c]*a_src[h,c]; same ald.
// ---------------------------------------------------------------------------
template <int HEADS, int CH>
__global__ void k_att_terms(const float* __restrict__ Hf, const float* __restrict__ asrc,
                            const float* __restrict__ adst, float* __restrict__ als,
                            float* __restrict__ ald, int n) {
  int t = blockIdx.x * blockDim.x + threadIdx.x;
  if (t >= n * HEADS) return;
  int hh = t & (HEADS - 1), i = t / HEADS;
  const float* hp = Hf + (size_t)i * HEADS * CH + (size_t)hh * CH;
  const float* as = asrc + hh * CH;
  const float* ad = adst + hh * CH;
  float s = 0.f, d = 0.f;
#pragma unroll 8
  for (int c = 0; c < CH; ++c) { float v = hp[c]; s += v * as[c]; d += v * ad[c]; }
  als[t] = s;
  ald[t] = d;
}

// ---------------------------------------------------------------------------
// edge pass 1: score = leakyReLU(als[src]+ald[dst]); atomicMax per-dst.
// self-loops are edges e >= E0 with src = dst = e - E0.
// ---------------------------------------------------------------------------
template <int HEADS>
__global__ void k_edge_score_max(const float* __restrict__ als, const float* __restrict__ ald,
                                 const int* __restrict__ esrc, const int* __restrict__ edst,
                                 float* __restrict__ sc, unsigned* __restrict__ mkey,
                                 int E0, int Etot) {
  int t = blockIdx.x * blockDim.x + threadIdx.x;
  if (t >= Etot * HEADS) return;
  int hh = t & (HEADS - 1), e = t / HEADS;
  int s = (e < E0) ? esrc[e] : (e - E0);
  int d = (e < E0) ? edst[e] : (e - E0);
  float v = als[s * HEADS + hh] + ald[d * HEADS + hh];
  v = (v > 0.f) ? v : 0.2f * v;
  sc[t] = v;
  atomicMax(&mkey[d * HEADS + hh], fkey(v));
}

// edge pass 2: ex = exp(score - max[dst]); accumulate per-dst denominator.
template <int HEADS>
__global__ void k_edge_exp_sum(const int* __restrict__ edst, float* __restrict__ sc,
                               const unsigned* __restrict__ mkey, float* __restrict__ den,
                               int E0, int Etot) {
  int t = blockIdx.x * blockDim.x + threadIdx.x;
  if (t >= Etot * HEADS) return;
  int hh = t & (HEADS - 1), e = t / HEADS;
  int d = (e < E0) ? edst[e] : (e - E0);
  float m = unfkey(mkey[d * HEADS + hh]);
  float ex = expf(sc[t] - m);
  sc[t] = ex;
  atomicAdd(&den[d * HEADS + hh], ex);
}

// edge pass 3: wave per edge; 32 lanes sweep all HEADS*CH channels,
// atomic-adding alpha-weighted source features into out[dst]. L2-resident.
template <int HEADS, int CH>
__global__ void k_edge_scatter(const float* __restrict__ Hf, const float* __restrict__ sc,
                               const float* __restrict__ den,
                               const int* __restrict__ esrc, const int* __restrict__ edst,
                               float* __restrict__ out, int E0, int Etot) {
  const int F = HEADS * CH;
  int wave = (int)((blockIdx.x * (long)blockDim.x + threadIdx.x) >> 5);
  int lane = threadIdx.x & 31;
  if (wave >= Etot) return;
  int e = wave;
  int s = (e < E0) ? esrc[e] : (e - E0);
  int d = (e < E0) ? edst[e] : (e - E0);
  const float* hp = Hf + (size_t)s * F;
  float* op = out + (size_t)d * F;
  const float* sce = sc + (size_t)e * HEADS;
  const float* dend = den + (size_t)d * HEADS;
  float alpha[HEADS];
#pragma unroll
  for (int hh = 0; hh < HEADS; ++hh) alpha[hh] = sce[hh] / dend[hh];
#pragma unroll
  for (int k = 0; k < F / 32; ++k) {
    int j = k * 32 + lane;
    atomicAdd(&op[j], hp[j] * alpha[j / CH]);   // j/CH is compile-time per k
  }
}

// out = aggregated + bias, ELU, convert to f16 for next GEMM
template <int F>  // F power of two (256)
__global__ void k_elu_bias_cvt(const float* __restrict__ o, const float* __restrict__ b,
                               _Float16* __restrict__ h, long total) {
  long i = (long)blockIdx.x * blockDim.x + threadIdx.x;
  if (i >= total) return;
  float v = o[i] + b[(int)i & (F - 1)];
  v = (v > 0.f) ? v : (expf(v) - 1.f);
  h[i] = (_Float16)v;
}

// ---------------------------------------------------------------------------
extern "C" void kernel_launch(void* const* d_in, const int* in_sizes, int n_in,
                              void* d_out, int out_size, void* d_ws, size_t ws_size,
                              hipStream_t stream) {
  const float* x  = (const float*)d_in[0];
  const float* W1 = (const float*)d_in[1];
  const float* as1 = (const float*)d_in[2];
  const float* ad1 = (const float*)d_in[3];
  const float* b1 = (const float*)d_in[4];
  const float* W2 = (const float*)d_in[5];
  const float* as2 = (const float*)d_in[6];
  const float* ad2 = (const float*)d_in[7];
  const float* b2 = (const float*)d_in[8];
  const int* esrc = (const int*)d_in[9];
  const int* edst = (const int*)d_in[10];
  float* out = (float*)d_out;

  const int IN = 128, F1 = 256, H1 = 4, C1 = 64, OUT = 64;
  const int N = in_sizes[0] / IN;
  const int E0 = in_sizes[9];
  const int Etot = E0 + N;

  size_t off = 0;
  auto alloc = [&](size_t bytes) -> void* {
    off = (off + 255) & ~(size_t)255;
    void* p = (char*)d_ws + off;
    off += bytes;
    return p;
  };
  _Float16* xh  = (_Float16*)alloc((size_t)N * IN * 2);
  _Float16* w1p = (_Float16*)alloc((size_t)IN * F1 * 2);
  _Float16* w2p = (_Float16*)alloc((size_t)F1 * OUT * 2);
  float*    h1  = (float*)alloc((size_t)N * F1 * 4);
  float*    als1 = (float*)alloc((size_t)N * H1 * 4);
  float*    ald1 = (float*)alloc((size_t)N * H1 * 4);
  unsigned* m1  = (unsigned*)alloc((size_t)N * H1 * 4);
  float*    den1 = (float*)alloc((size_t)N * H1 * 4);
  float*    sc1 = (float*)alloc((size_t)Etot * H1 * 4);
  float*    o1  = (float*)alloc((size_t)N * F1 * 4);
  _Float16* h2h = (_Float16*)alloc((size_t)N * F1 * 2);
  // layer-2 buffers alias freed layer-1 scratch
  float* g2 = h1;                 // [N,64] fits in h1's [N,256] region
  float* als2 = als1; float* ald2 = ald1;
  unsigned* m2 = m1;  float* den2 = den1;
  float* sc2 = sc1;
  (void)ws_size; (void)n_in; (void)out_size;

  const int B = 256;
  auto blks = [](long n, int b) { return (int)((n + b - 1) / b); };
  int mtiles = (N + 15) / 16;

  // ---- layer 1 ----
  k_cvt_f16<<<blks((long)N * IN, B), B, 0, stream>>>(x, xh, (long)N * IN);
  k_prepack_w<<<blks(IN * F1, B), B, 0, stream>>>(W1, w1p, IN, F1);
  k_prepack_w<<<blks(F1 * OUT, B), B, 0, stream>>>(W2, w2p, F1, OUT);

  k_gemm_wmma<4, 16><<<blks((long)mtiles * 32, B), B, 0, stream>>>(xh, w1p, h1, N);
  k_att_terms<4, 64><<<blks((long)N * H1, B), B, 0, stream>>>(h1, as1, ad1, als1, ald1, N);
  k_fill_u32<<<blks(N * H1, B), B, 0, stream>>>(m1, NEG_INF_KEY, N * H1);
  k_fill_f32<<<blks((long)N * H1, B), B, 0, stream>>>(den1, 0.f, (long)N * H1);
  k_fill_f32<<<blks((long)N * F1, B), B, 0, stream>>>(o1, 0.f, (long)N * F1);
  k_edge_score_max<4><<<blks((long)Etot * H1, B), B, 0, stream>>>(als1, ald1, esrc, edst, sc1, m1, E0, Etot);
  k_edge_exp_sum<4><<<blks((long)Etot * H1, B), B, 0, stream>>>(edst, sc1, m1, den1, E0, Etot);
  k_edge_scatter<4, 64><<<blks((long)Etot * 32, B), B, 0, stream>>>(h1, sc1, den1, esrc, edst, o1, E0, Etot);
  k_elu_bias_cvt<256><<<blks((long)N * F1, B), B, 0, stream>>>(o1, b1, h2h, (long)N * F1);

  // ---- layer 2 ----
  k_gemm_wmma<8, 4><<<blks((long)mtiles * 32, B), B, 0, stream>>>(h2h, w2p, g2, N);
  k_att_terms<1, 64><<<blks((long)N, B), B, 0, stream>>>(g2, as2, ad2, als2, ald2, N);
  k_fill_u32<<<blks(N, B), B, 0, stream>>>(m2, NEG_INF_KEY, N);
  k_fill_f32<<<blks((long)N, B), B, 0, stream>>>(den2, 0.f, (long)N);
  k_bias_rows<64><<<blks((long)N * OUT, B), B, 0, stream>>>(out, b2, N);
  k_edge_score_max<1><<<blks((long)Etot, B), B, 0, stream>>>(als2, ald2, esrc, edst, sc2, m2, E0, Etot);
  k_edge_exp_sum<1><<<blks((long)Etot, B), B, 0, stream>>>(edst, sc2, m2, den2, E0, Etot);
  k_edge_scatter<1, 64><<<blks((long)Etot * 32, B), B, 0, stream>>>(g2, sc2, den2, esrc, edst, out, E0, Etot);
}
